// EghnQNet_38448547234264
// MI455X (gfx1250) — matrix operand order repaired and checked by reading, not compiled, not run on hardware
//
#include <hip/hip_runtime.h>
#include <hip/hip_bf16.h>

// =====================================================================
// EGHN Q-network forward, fully fused: one workgroup (8 wave32s) per
// graph, all intermediates in LDS, all GEMMs through
// v_wmma_f32_16x16x32_bf16 (bf16 operands, fp32 accumulate).
// GEMM output buffers are padded to full 16-row tiles and invalid-edge
// atomics are routed to a trash row so all WMMA epilogues are
// branch-free straight-line code.
// =====================================================================

#define NT 256  // threads per block = 8 waves (wave32)

typedef __bf16 bf16;
typedef __attribute__((ext_vector_type(16))) __bf16 bf16x16;
typedef __attribute__((ext_vector_type(8)))  __bf16 bf16x8;
typedef __attribute__((ext_vector_type(8)))  float  v8f;

struct EgnnP {
  const float *e0b,*e0w,*e1b,*e1w;   // phi_e: [258->128, 128->128]
  const float *h0b,*h0w,*h1b,*h1w;   // phi_h: [256->128, 128->128]
  const float *vb,*vw;               // phi_v: [128->1]
  const float *x0b,*x0w,*x1b,*x1w;   // phi_x: [128->128, 128->1]
};

struct NetArgs {
  const float *cent, *act;
  const float *crit_b, *crit_w;
  const float *dec0b,*dec0w,*dec1b,*dec1w;
  const float *emb_b,*emb_w;
  EgnnP high0, low0, low1;
  const float *pool0b,*pool0w,*pool1b,*pool1w;
  float *out;
};

struct Smem {
  alignas(64) bf16 Wt[128*256];   // weights, transposed: Wt[n*256 + k] (2 x 128-K slots or 1 x 256-K)
  alignas(64) bf16 A[64*128];     // GEMM A staging (m1 chunk / 32x256 concat)
  alignas(64) bf16 M[64*128];     // GEMM A staging (m chunk / hidden)
  alignas(64) bf16 hb[32*128];    // node features bf16, zero-padded rows
  alignas(16) float hN[32*128];   // node features fp32 (rows 20..31 = pad/garbage)
  float Hc[16*128];               // cluster features fp32 (rows 4..15 = pad/garbage)
  float P[32*128], Q[32*128];     // phi_e layer-0 node pre-projections (padded)
  float aggm[32*128];             // segment-sum of m; row nn = trash row
  float hback[20*128];
  float w_d2[128], w_ea[128], pxw[128];
  float eattr[380], ceattr[12];
  float x[40], v[40], Xc[8], Vc[8], aggx[40], phv[20];
  float xij0[64], xij1[64], d2v[64], sdot[64];
  float s[80], sden[4], qbuf[20];
  int   ei[64], ej[64];
};
static_assert(sizeof(Smem) <= 320*1024, "over WGP LDS budget");

static __device__ __forceinline__ float silu_f(float x) { return x / (1.0f + __expf(-x)); }

static __device__ __forceinline__ v8f zero_acc() {
  v8f a;
#pragma unroll
  for (int r = 0; r < 8; ++r) a[r] = 0.0f;
  return a;
}

static __device__ __forceinline__ v8f bias_acc(const float* __restrict__ b, int nt) {
  float bv = b[nt*16 + (threadIdx.x & 15)];
  v8f a;
#pragma unroll
  for (int r = 0; r < 8; ++r) a[r] = bv;
  return a;
}

// Distributed prefetch of a weight blob (L2/WGP warm); lowers to global_prefetch_b8.
static __device__ __forceinline__ void prefetch_w(const float* __restrict__ W, int bytes) {
  for (int off = threadIdx.x*4096; off < bytes; off += NT*4096)
    __builtin_prefetch((const char*)W + off, 0, 1);
}

// One 16x16 C tile: C += A[16xK] * W[Kx16].
// A (LDS bf16, row-major, stride lda): lane = row (mod 16); per 32-K step the
// lane's 16 values are K = kb+g*16..+15  (g = lane/16), i.e.
// two contiguous 16B LDS loads, matching the B operand's K order.
// W staged transposed in LDS (Wt[n*256+k]): lane = col, 16 contiguous K
// values = one 32B LDS load.
static __device__ __forceinline__ v8f tile_mm(const bf16* __restrict__ Alds, int lda, int mt,
                                              const bf16* __restrict__ Wt, int nt,
                                              int K, v8f acc) {
  const int lane = threadIdx.x & 31;
  const int g = lane >> 4, rr = lane & 15;
  const bf16* arow = Alds + (mt*16 + rr) * lda;
  const bf16* brow = Wt + (nt*16 + rr) * 256;
  for (int kb = 0; kb < K; kb += 32) {
    union { bf16x16 v; bf16x8 h[2]; } a;
    a.h[0] = *(const bf16x8*)(arow + kb + g*16);
    a.h[1] = *(const bf16x8*)(arow + kb + g*16 + 8);
    bf16x16 b = *(const bf16x16*)(brow + kb + g*16);
    acc = __builtin_amdgcn_wmma_f32_16x16x32_bf16(false, a.v, false, b, (short)0, acc, false, false);
  }
  return acc;
}

// C/D layout: VGPR r, lanes 0-15 -> M = r, lanes 16-31 -> M = 8+r; N = lane&15.
// Unguarded: destination buffers are padded to full tiles.
static __device__ __forceinline__ void store_f32(float* __restrict__ C, int ldc, int mt, int nt,
                                                 v8f acc) {
  const int lane = threadIdx.x & 31;
  const int g = lane >> 4, n = nt*16 + (lane & 15);
#pragma unroll
  for (int r = 0; r < 8; ++r) {
    int m = mt*16 + g*8 + r;
    C[m*ldc + n] = acc[r];
  }
}

// Stage global f32 weight W[KxN=128] into LDS transposed bf16 Wt[n*256 + k].
static __device__ __forceinline__ void stage_wt(bf16* __restrict__ Wt, const float* __restrict__ W, int K) {
  for (int idx = threadIdx.x; idx < K*128; idx += NT) {
    int k = idx >> 7, n = idx & 127;               // idx order => coalesced global reads
    Wt[n*256 + k] = (bf16)W[idx];
  }
}

// h = h + MLP_silu( [h || other] ; W0(256x128)+b0, W1(128x128)+b1 ), refresh hb.
// hf must be padded to Mtn*16 rows (pad rows hold garbage, never read).
static __device__ __forceinline__ void residual_concat_mlp(float* __restrict__ hf, bf16* __restrict__ hb,
                                                           const float* __restrict__ other,
                                                           const float* b0, const float* W0,
                                                           const float* b1, const float* W1,
                                                           int nn, Smem& S) {
  const int tid = threadIdx.x, wave = tid >> 5;
  const int Mtn = (nn + 15) >> 4;
  for (int idx = tid; idx < Mtn*16*256; idx += NT) {
    int m = idx >> 8, k = idx & 255;
    float vl = 0.0f;
    if (m < nn) vl = (k < 128) ? hf[m*128 + k] : other[m*128 + (k - 128)];
    S.A[idx] = (bf16)vl;
  }
  stage_wt(S.Wt, W0, 256);
  prefetch_w(W1, 128*128*4);
  __syncthreads();
  for (int t = wave; t < Mtn*8; t += 8) {
    int mt = t >> 3, nt = t & 7;
    v8f acc = bias_acc(b0, nt);
    acc = tile_mm(S.A, 256, mt, S.Wt, nt, 256, acc);
    int lane = tid & 31, gg = lane >> 4, n = nt*16 + (lane & 15);
#pragma unroll
    for (int r = 0; r < 8; ++r) { int m = mt*16 + gg*8 + r; S.M[m*128 + n] = (bf16)silu_f(acc[r]); }
  }
  __syncthreads();
  stage_wt(S.Wt, W1, 128);
  __syncthreads();
  for (int t = wave; t < Mtn*8; t += 8) {
    int mt = t >> 3, nt = t & 7;
    v8f acc = bias_acc(b1, nt);
    acc = tile_mm(S.M, 128, mt, S.Wt, nt, 128, acc);
    int lane = tid & 31, gg = lane >> 4, n = nt*16 + (lane & 15);
#pragma unroll
    for (int r = 0; r < 8; ++r) { int m = mt*16 + gg*8 + r; hf[m*128 + n] += acc[r]; }
  }
  __syncthreads();
  for (int idx = tid; idx < nn*128; idx += NT) hb[idx] = (bf16)hf[idx];
  __syncthreads();
}

// One EGNN layer on nn fully-connected nodes (nn=20 low, nn=4 high).
static __device__ __forceinline__ void egnn_layer(const EgnnP& p, int nn,
                                                  float* __restrict__ hf, bf16* __restrict__ hb,
                                                  float* __restrict__ x, float* __restrict__ v,
                                                  const float* __restrict__ eattr, Smem& S) {
  const int tid = threadIdx.x, wave = tid >> 5;
  const int E = nn*(nn-1), deg = nn - 1;
  const int Mtn = (nn + 15) >> 4;

  // phi_e layer-0 split: P = h@W[0:128], Q = h@W[128:256]; rows 256/257 = d2/eattr weights.
  for (int c = tid; c < 128; c += NT) {
    S.w_d2[c] = p.e0w[256*128 + c];
    S.w_ea[c] = p.e0w[257*128 + c];
    S.pxw[c]  = p.x1w[c];
  }
  stage_wt(S.Wt,       p.e0w,           128);   // slot0: W1a
  stage_wt(S.Wt + 128, p.e0w + 128*128, 128);   // slot1: W1b
  for (int idx = tid; idx < nn*128; idx += NT) S.aggm[idx] = 0.0f;
  for (int idx = nn*128 + tid; idx < Mtn*16*128; idx += NT) hb[idx] = (bf16)0.0f;  // zero pad rows
  if (tid < nn*2) S.aggx[tid] = 0.0f;
  prefetch_w(p.e1w, 128*128*4);
  prefetch_w(p.x0w, 128*128*4);
  __syncthreads();

  for (int t = wave; t < Mtn*8; t += 8) {
    int mt = t >> 3, nt = t & 7;
    v8f aP = zero_acc(); aP = tile_mm(hb, 128, mt, S.Wt,       nt, 128, aP); store_f32(S.P, 128, mt, nt, aP);
    v8f aQ = zero_acc(); aQ = tile_mm(hb, 128, mt, S.Wt + 128, nt, 128, aQ); store_f32(S.Q, 128, mt, nt, aQ);
  }
  if (tid < nn) {  // phi_v(h) scalar gate (uses pre-update h)
    float a = p.vb[0];
    for (int c = 0; c < 128; ++c) a += hf[tid*128 + c] * p.vw[c];
    S.phv[tid] = a;
  }
  __syncthreads();

  stage_wt(S.Wt,       p.e1w, 128);   // slot0: phi_e layer-1
  stage_wt(S.Wt + 128, p.x0w, 128);   // slot1: phi_x layer-0
  prefetch_w(p.h0w, 256*128*4);
  __syncthreads();

  const int nch = (E + 63) >> 6;
  for (int c = 0; c < nch; ++c) {
    const int eb = c*64;
    if (tid < 64) {  // per-edge geometry for this chunk
      int e = eb + tid, i = nn, j = nn;  // invalid edges -> trash row nn
      float xx = 0.0f, yy = 0.0f;
      if (e < E) {
        i = e / deg; int jj = e - i*deg; j = jj + (jj >= i);
        xx = x[i*2] - x[j*2]; yy = x[i*2+1] - x[j*2+1];
      }
      S.ei[tid] = i; S.ej[tid] = j;
      S.xij0[tid] = xx; S.xij1[tid] = yy; S.d2v[tid] = xx*xx + yy*yy;
      S.sdot[tid] = 0.0f;
    }
    __syncthreads();
    // m1 = silu(P[i] + Q[j] + d2*w_d2 + eattr*w_ea + b0)
    for (int idx = tid; idx < 64*128; idx += NT) {
      int el = idx >> 7, ch = idx & 127, e = eb + el;
      float vl = 0.0f;
      if (e < E) {
        vl = S.P[S.ei[el]*128 + ch] + S.Q[S.ej[el]*128 + ch]
           + S.d2v[el]*S.w_d2[ch] + eattr[e]*S.w_ea[ch] + p.e0b[ch];
        vl = silu_f(vl);
      }
      S.A[idx] = (bf16)vl;
    }
    __syncthreads();
    // m = silu(m1 @ W2 + b2), fused agg_m segment-sum from accumulators.
    // Invalid edges have ei == nn -> their adds land in the trash row.
    for (int t = wave; t < 32; t += 8) {
      int mt = t >> 3, nt = t & 7;
      v8f acc = bias_acc(p.e1b, nt);
      acc = tile_mm(S.A, 128, mt, S.Wt, nt, 128, acc);
      int lane = tid & 31, gg = lane >> 4, n = nt*16 + (lane & 15);
#pragma unroll
      for (int r = 0; r < 8; ++r) {
        int m = mt*16 + gg*8 + r;
        float vl = silu_f(acc[r]);
        S.M[m*128 + n] = (bf16)vl;
        atomicAdd(&S.aggm[S.ei[m]*128 + n], vl);
      }
    }
    __syncthreads();
    // t = silu(m @ W3 + b3); fused dot with phi_x layer-1 weight (128->1)
    for (int t = wave; t < 32; t += 8) {
      int mt = t >> 3, nt = t & 7;
      v8f acc = bias_acc(p.x0b, nt);
      acc = tile_mm(S.M, 128, mt, S.Wt + 128, nt, 128, acc);
      int lane = tid & 31, gg = lane >> 4, n = nt*16 + (lane & 15);
      float w4 = S.pxw[n];
      float part[8];
#pragma unroll
      for (int r = 0; r < 8; ++r) part[r] = silu_f(acc[r]) * w4;
#pragma unroll
      for (int msk = 1; msk < 16; msk <<= 1) {
#pragma unroll
        for (int r = 0; r < 8; ++r) part[r] += __shfl_xor(part[r], msk, 32);
      }
      if ((lane & 15) == 0) {
#pragma unroll
        for (int r = 0; r < 8; ++r) atomicAdd(&S.sdot[mt*16 + gg*8 + r], part[r]);
      }
    }
    __syncthreads();
    if (tid < 64) {  // agg_x += xij * phi_x(m)
      int e = eb + tid;
      if (e < E) {
        float sc = S.sdot[tid] + p.x1b[0];
        atomicAdd(&S.aggx[S.ei[tid]*2],     S.xij0[tid]*sc);
        atomicAdd(&S.aggx[S.ei[tid]*2 + 1], S.xij1[tid]*sc);
      }
    }
    __syncthreads();
  }

  if (tid < nn) {  // v = phi_v(h)*v + agg_x/deg ; x += v
    float pv = S.phv[tid], inv = 1.0f / (float)deg;
#pragma unroll
    for (int d = 0; d < 2; ++d) {
      float nv = pv*v[tid*2 + d] + S.aggx[tid*2 + d]*inv;
      v[tid*2 + d] = nv;
      x[tid*2 + d] += nv;
    }
  }
  __syncthreads();

  // h = h + phi_h([h || agg_m])
  residual_concat_mlp(hf, hb, S.aggm, p.h0b, p.h0w, p.h1b, p.h1w, nn, S);
}

__global__ __launch_bounds__(NT, 1) void eghn_qnet_kernel(NetArgs Ar) {
  extern __shared__ char smem_raw[];
  Smem& S = *reinterpret_cast<Smem*>(smem_raw);
  const int g = blockIdx.x, tid = threadIdx.x, wave = tid >> 5;

  // ---- embedding + initial x/v ----
  for (int idx = tid; idx < 20*128; idx += NT) {
    int n = idx >> 7, ch = idx & 127;
    const float* row = Ar.cent + (g*20 + n)*14;
    float a = Ar.emb_b[ch];
#pragma unroll
    for (int k = 0; k < 10; ++k) a += row[k]*Ar.emb_w[k*128 + ch];
    S.hN[idx] = a; S.hb[idx] = (bf16)a;
  }
  if (tid < 20) {
    const float* row = Ar.cent + (g*20 + tid)*14;
    S.x[tid*2] = row[10]; S.x[tid*2+1] = row[11];
    S.v[tid*2] = Ar.act[(g*20 + tid)*2]; S.v[tid*2+1] = Ar.act[(g*20 + tid)*2 + 1];
  }
  __syncthreads();
  for (int e = tid; e < 380; e += NT) {  // eattr from initial loc (constant across layers)
    int i = e / 19, jj = e - i*19, j = jj + (jj >= i);
    float dx = S.x[i*2] - S.x[j*2], dy = S.x[i*2+1] - S.x[j*2+1];
    S.eattr[e] = dx*dx + dy*dy;
  }
  __syncthreads();

  // ---- low EGNN layers ----
  egnn_layer(Ar.low0, 20, S.hN, S.hb, S.x, S.v, S.eattr, S);
  egnn_layer(Ar.low1, 20, S.hN, S.hb, S.x, S.v, S.eattr, S);

  // ---- pooling: s = softmax(MLP(h)) ----
  stage_wt(S.Wt, Ar.pool0w, 128);
  __syncthreads();
  for (int t = wave; t < 16; t += 8) {
    int mt = t >> 3, nt = t & 7;
    v8f acc = bias_acc(Ar.pool0b, nt);
    acc = tile_mm(S.hb, 128, mt, S.Wt, nt, 128, acc);
    int lane = tid & 31, gg = lane >> 4, n = nt*16 + (lane & 15);
#pragma unroll
    for (int r = 0; r < 8; ++r) { int m = mt*16 + gg*8 + r; S.M[m*128 + n] = (bf16)silu_f(acc[r]); }
  }
  __syncthreads();
  if (tid < 20) {
    float lg[4];
#pragma unroll
    for (int k = 0; k < 4; ++k) {
      float a = Ar.pool1b[k];
      for (int ch = 0; ch < 128; ++ch) a += (float)S.M[tid*128 + ch]*Ar.pool1w[ch*4 + k];
      lg[k] = a;
    }
    float mx = fmaxf(fmaxf(lg[0], lg[1]), fmaxf(lg[2], lg[3]));
    float e0 = __expf(lg[0]-mx), e1 = __expf(lg[1]-mx), e2 = __expf(lg[2]-mx), e3 = __expf(lg[3]-mx);
    float sm = e0 + e1 + e2 + e3;
    S.s[tid*4+0] = e0/sm; S.s[tid*4+1] = e1/sm; S.s[tid*4+2] = e2/sm; S.s[tid*4+3] = e3/sm;
  }
  __syncthreads();
  if (tid < 4) { float a = 1e-6f; for (int n = 0; n < 20; ++n) a += S.s[n*4 + tid]; S.sden[tid] = a; }
  __syncthreads();
  for (int idx = tid; idx < 4*128; idx += NT) {  // Hc
    int k = idx >> 7, ch = idx & 127;
    float a = 0.0f;
    for (int n = 0; n < 20; ++n) a += S.s[n*4 + k]*S.hN[n*128 + ch];
    S.Hc[idx] = a / S.sden[k];
  }
  if (tid < 16) {  // Xc, Vc
    int k = (tid >> 1) & 3, d = tid & 1;
    bool isv = tid >= 8;
    float a = 0.0f;
    for (int n = 0; n < 20; ++n) a += S.s[n*4 + k]*(isv ? S.v[n*2 + d] : S.x[n*2 + d]);
    a /= S.sden[k];
    if (isv) S.Vc[k*2 + d] = a; else S.Xc[k*2 + d] = a;
  }
  __syncthreads();
  if (tid < 12) {  // cluster edge attributes
    int i = tid / 3, jj = tid - i*3, j = jj + (jj >= i);
    float dx = S.Xc[i*2] - S.Xc[j*2], dy = S.Xc[i*2+1] - S.Xc[j*2+1];
    S.ceattr[tid] = dx*dx + dy*dy;
  }
  for (int idx = tid; idx < 4*128; idx += NT) S.hb[idx] = (bf16)S.Hc[idx];
  __syncthreads();

  // ---- high EGNN layer on clusters ----
  egnn_layer(Ar.high0, 4, S.Hc, S.hb, S.Xc, S.Vc, S.ceattr, S);

  // ---- hback = s @ Hc ----
  for (int idx = tid; idx < 20*128; idx += NT) {
    int n = idx >> 7, ch = idx & 127;
    float a = 0.0f;
#pragma unroll
    for (int k = 0; k < 4; ++k) a += S.s[n*4 + k]*S.Hc[k*128 + ch];
    S.hback[idx] = a;
  }
  __syncthreads();

  // ---- decoder: h = h + MLP([h || hback]) ----
  residual_concat_mlp(S.hN, S.hb, S.hback, Ar.dec0b, Ar.dec0w, Ar.dec1b, Ar.dec1w, 20, S);

  // ---- critic + mean over nodes ----
  if (tid < 20) {
    float a = Ar.crit_b[0];
    for (int ch = 0; ch < 128; ++ch) a += tanhf(S.hN[tid*128 + ch])*Ar.crit_w[ch];
    S.qbuf[tid] = a;
  }
  __syncthreads();
  if (tid == 0) {
    float a = 0.0f;
    for (int n = 0; n < 20; ++n) a += S.qbuf[n];
    Ar.out[g] = a*0.05f;
  }
}

// =====================================================================
// Host launch. Parameter pointers follow jax tree_flatten of the
// setup_inputs() dict: cent_obs, actions, then params with dict keys
// sorted (critic, dec, embed, high, low, pool; leaves 'b' before 'w';
// egnn dicts: phi_e, phi_h, phi_v, phi_x).
// =====================================================================
extern "C" void kernel_launch(void* const* d_in, const int* in_sizes, int n_in,
                              void* d_out, int out_size, void* d_ws, size_t ws_size,
                              hipStream_t stream) {
  (void)n_in; (void)d_ws; (void)ws_size; (void)out_size;
  auto F = [&](int i) { return (const float*)d_in[i]; };
  auto EG = [&](int b) {
    EgnnP p;
    p.e0b = F(b+0);  p.e0w = F(b+1);  p.e1b = F(b+2);  p.e1w = F(b+3);
    p.h0b = F(b+4);  p.h0w = F(b+5);  p.h1b = F(b+6);  p.h1w = F(b+7);
    p.vb  = F(b+8);  p.vw  = F(b+9);
    p.x0b = F(b+10); p.x0w = F(b+11); p.x1b = F(b+12); p.x1w = F(b+13);
    return p;
  };

  NetArgs a;
  a.cent = F(0); a.act = F(1);
  a.crit_b = F(2); a.crit_w = F(3);
  a.dec0b = F(4); a.dec0w = F(5); a.dec1b = F(6); a.dec1w = F(7);
  a.emb_b = F(8); a.emb_w = F(9);
  a.high0 = EG(10);
  a.low0  = EG(24);
  a.low1  = EG(38);
  a.pool0b = F(52); a.pool0w = F(53); a.pool1b = F(54); a.pool1w = F(55);
  a.out = (float*)d_out;

  const int B = in_sizes[0] / (20*14);  // 1024 graphs

  (void)hipFuncSetAttribute((const void*)eghn_qnet_kernel,
                            hipFuncAttributeMaxDynamicSharedMemorySize,
                            (int)sizeof(Smem));
  eghn_qnet_kernel<<<dim3(B), dim3(NT), sizeof(Smem), stream>>>(a);
}
